// T5ScaledDotProductAttention_46205258170724
// MI455X (gfx1250) — compile-verified, matrix operand order checked by reading, not run on hardware
//
#include <hip/hip_runtime.h>
#include <hip/hip_bf16.h>

// ---- problem constants (T5 attention reference) ----
#define BH 32
#define SEQ 2048
#define DIM 128

typedef __attribute__((ext_vector_type(16))) __bf16 v16bf;
typedef __attribute__((ext_vector_type(8)))  __bf16 bf16x8;
typedef __attribute__((ext_vector_type(4)))  __bf16 bf16x4;
typedef __attribute__((ext_vector_type(8)))  float  v8f;
typedef __attribute__((ext_vector_type(4)))  float  f32x4;

#define KSTRIDE 136   // klds row stride: 272B = 17*16B, banks r*4  mod 64 distinct
#define VSTRIDE 56    // vt   row stride: 112B = 7*16B,  banks r*28 mod 64 distinct
#define PSTRIDE 36    // pf32 row stride: 144B = 9*16B,  banks r*36 mod 64 distinct

// ---- DPP16 butterfly reductions within each 16-lane half (wave32) ----
__device__ __forceinline__ float dpp_max16(float x) {
    int t;
    t = __builtin_amdgcn_update_dpp(__float_as_int(x), __float_as_int(x), 0xB1,  0xF, 0xF, false); // quad xor1
    x = fmaxf(x, __int_as_float(t));
    t = __builtin_amdgcn_update_dpp(__float_as_int(x), __float_as_int(x), 0x4E,  0xF, 0xF, false); // quad xor2
    x = fmaxf(x, __int_as_float(t));
    t = __builtin_amdgcn_update_dpp(__float_as_int(x), __float_as_int(x), 0x141, 0xF, 0xF, false); // half mirror
    x = fmaxf(x, __int_as_float(t));
    t = __builtin_amdgcn_update_dpp(__float_as_int(x), __float_as_int(x), 0x140, 0xF, 0xF, false); // row mirror
    x = fmaxf(x, __int_as_float(t));
    return x;
}
__device__ __forceinline__ float dpp_sum16(float x) {
    int t;
    t = __builtin_amdgcn_update_dpp(__float_as_int(x), __float_as_int(x), 0xB1,  0xF, 0xF, false);
    x += __int_as_float(t);
    t = __builtin_amdgcn_update_dpp(__float_as_int(x), __float_as_int(x), 0x4E,  0xF, 0xF, false);
    x += __int_as_float(t);
    t = __builtin_amdgcn_update_dpp(__float_as_int(x), __float_as_int(x), 0x141, 0xF, 0xF, false);
    x += __int_as_float(t);
    t = __builtin_amdgcn_update_dpp(__float_as_int(x), __float_as_int(x), 0x140, 0xF, 0xF, false);
    x += __int_as_float(t);
    return x;
}

__global__ __launch_bounds__(256)
void t5_attn_kernel(const float* __restrict__ Q, const float* __restrict__ K,
                    const float* __restrict__ V, const float* __restrict__ Bias,
                    const unsigned char* __restrict__ Mask,
                    float* __restrict__ Out, float* __restrict__ Attn)
{
    constexpr float NEGV  = -1e18f;
    constexpr float L2E   = 1.44269504088896340736f;
    constexpr float SCALE = 0.08838834764831845f;   // 1/sqrt(128)
    constexpr int   NKT   = SEQ / 32;               // 64 key-tile iterations

    __shared__ __align__(16) __bf16 klds[2][32][KSTRIDE];  // K tile, row-major bf16
    __shared__ __align__(16) __bf16 vt[2][DIM][VSTRIDE];   // V tile, transposed bf16
    __shared__ __align__(16) float  pf32[8][16][PSTRIDE];  // per-wave p tile, f32
    __shared__ float invlds[128];                          // 1/rowsum broadcast

    const int tid  = threadIdx.x;
    const int wave = tid >> 5;
    const int lane = tid & 31;
    const int half = lane >> 4;   // K-half selector for A/B fragments
    const int l16  = lane & 15;

    const int bh    = blockIdx.x >> 4;          // 16 q-tiles of 128 per head
    const int qtile = blockIdx.x & 15;
    const int q0    = qtile * 128 + wave * 16;  // this wave's 16 q rows

    const float* Qb = Q + (size_t)bh * SEQ * DIM;
    const float* Kb = K + (size_t)bh * SEQ * DIM;
    const float* Vb = V + (size_t)bh * SEQ * DIM;

    // per-lane base pointers; j/t parts become immediate IOFFSETs (24-bit)
    const int qbase = q0 + 8 * half;                   // row of j==0
    const float* bbase = Bias
        + ((size_t)qbase * SEQ + l16) * BH + bh;       // + j*SEQ*BH + t*16*BH
    const unsigned char* mbase = Mask
        + ((size_t)bh * SEQ + qbase) * SEQ + l16;      // + j*SEQ + t*16
    float* awave = Attn
        + ((size_t)bh * SEQ + q0) * SEQ;               // + row*SEQ + k0 + col

    // staging assignment: thread stages one key row segment of 16 floats
    const int kk  = tid >> 3;          // 0..31  key row within tile
    const int dd0 = (tid & 7) * 16;    // 0..112 column base

    f32x4 kreg[4], vreg[4];            // in-flight next tile
    auto issue_loads = [&](int k0) {
        const float* kp = Kb + (size_t)(k0 + kk) * DIM + dd0;
        const float* vp = Vb + (size_t)(k0 + kk) * DIM + dd0;
#pragma unroll
        for (int i = 0; i < 4; ++i) {
            kreg[i] = *(const f32x4*)(kp + 4 * i);
            vreg[i] = *(const f32x4*)(vp + 4 * i);
        }
    };
    auto store_tiles = [&](int buf) {
#pragma unroll
        for (int i = 0; i < 4; ++i) {
            bf16x4 tk;
#pragma unroll
            for (int j = 0; j < 4; ++j) tk[j] = (__bf16)kreg[i][j];
            *(bf16x4*)&klds[buf][kk][dd0 + 4 * i] = tk;     // 8B contiguous
#pragma unroll
            for (int j = 0; j < 4; ++j)                      // transposed scatter
                vt[buf][dd0 + 4 * i + j][kk] = (__bf16)vreg[i][j];
        }
    };

    // ---- load Q as 4 A-fragments (D=128 = 4 chunks of K=32), kept in VGPRs ----
    v16bf qa[4];
    {
        const float* qrow = Qb + (size_t)(q0 + l16) * DIM;
#pragma unroll
        for (int c = 0; c < 4; ++c) {
            const int base = c * 32 + half * 8;
            f32x4 f0 = *(const f32x4*)(qrow + base);
            f32x4 f1 = *(const f32x4*)(qrow + base + 4);
            f32x4 f2 = *(const f32x4*)(qrow + base + 16);
            f32x4 f3 = *(const f32x4*)(qrow + base + 20);
            v16bf a;
#pragma unroll
            for (int i = 0; i < 4; ++i) {
                a[i]      = (__bf16)f0[i];
                a[4 + i]  = (__bf16)f1[i];
                a[8 + i]  = (__bf16)f2[i];
                a[12 + i] = (__bf16)f3[i];
            }
            qa[c] = a;
        }
    }

    // ---- accumulators (r[] kept as per-lane PARTIAL sums; reduced at end) ----
    v8f oacc[8];
#pragma unroll
    for (int t = 0; t < 8; ++t) { v8f z = {}; oacc[t] = z; }
    float m[8], r[8];
#pragma unroll
    for (int j = 0; j < 8; ++j) { m[j] = -3.0e38f; r[j] = 0.0f; }

    // ---- pipeline prologue: stage tile 0 ----
    issue_loads(0);
    store_tiles(0);
    __syncthreads();

    // =================== main loop: 32 keys per iteration ===================
    for (int kt = 0; kt < NKT; ++kt) {
        const int buf = kt & 1;

        // issue next tile's global loads early (latency hidden under compute)
        if (kt + 1 < NKT) issue_loads(kt * 32 + 32);

        // hoist bias / mask gathers for this tile (immediate-offset loads)
        float    bi[2][8];
        unsigned mkbits = 0;
#pragma unroll
        for (int j = 0; j < 8; ++j) {
#pragma unroll
            for (int t = 0; t < 2; ++t) {
                bi[t][j] = bbase[(size_t)j * SEQ * BH + t * 16 * BH];
                mkbits |= (unsigned)mbase[j * SEQ + t * 16] << (t * 8 + j);
            }
            if (kt + 1 < NKT)   // prefetch next iteration's bias lines
                __builtin_prefetch(bbase + (size_t)j * SEQ * BH + 32 * BH, 0, 3);
        }

        // --- scores: two 16x16 tiles (32 keys), 8 WMMAs, B-frags from LDS ---
        v8f sacc[2];
#pragma unroll
        for (int t = 0; t < 2; ++t) {
            v8f z = {}; sacc[t] = z;
            const __bf16* krow = &klds[buf][t * 16 + l16][0];
#pragma unroll
            for (int c = 0; c < 4; ++c) {
                v16bf kb;
                *(bf16x8*)&kb       = *(const bf16x8*)(krow + c * 32 + half * 8);
                *((bf16x8*)&kb + 1) = *(const bf16x8*)(krow + c * 32 + half * 8 + 16);
                sacc[t] = __builtin_amdgcn_wmma_f32_16x16x32_bf16(
                    false, qa[c], false, kb, (short)0, sacc[t], false, false);
            }
        }

        // --- scale + bias + mask, per-row tile max (DPP16 butterfly) ---
        float mn[8];
        bool  upd = false;
#pragma unroll
        for (int j = 0; j < 8; ++j) {
            float tm = -3.0e38f;
#pragma unroll
            for (int t = 0; t < 2; ++t) {
                float s = sacc[t][j] * SCALE + bi[t][j];
                if (mkbits & (1u << (t * 8 + j))) s = NEGV;
                sacc[t][j] = s;
                tm = fmaxf(tm, s);
            }
            mn[j] = fmaxf(m[j], dpp_max16(tm));
            upd = upd || (mn[j] > m[j]);
        }

        // --- rescale only if any row max moved (wave-uniform branch) ---
        if (__ballot(upd)) {
#pragma unroll
            for (int j = 0; j < 8; ++j) {
                const float fs = __builtin_amdgcn_exp2f((m[j] - mn[j]) * L2E);
                r[j] *= fs;                       // per-lane partials scale too
#pragma unroll
                for (int dt = 0; dt < 8; ++dt)
                    oacc[dt][j] *= fs;
            }
        }
#pragma unroll
        for (int j = 0; j < 8; ++j) m[j] = mn[j];

        // --- p = exp(s - m): accumulate partials, stage f32 tile in LDS ---
#pragma unroll
        for (int j = 0; j < 8; ++j) {
            const float mj = m[j];
#pragma unroll
            for (int t = 0; t < 2; ++t) {
                const float p = __builtin_amdgcn_exp2f((sacc[t][j] - mj) * L2E);
                r[j] += p;                                  // per-lane partial
                pf32[wave][j + 8 * half][t * 16 + l16] = p;
            }
        }

        // --- write attn tile: 4 coalesced b128 stores per lane (128B rows) ---
#pragma unroll
        for (int k = 0; k < 4; ++k) {
            const int fid = lane + 32 * k;        // 128 float4 in 16x32 tile
            const int row = fid >> 3;
            const int c4  = fid & 7;
            f32x4 v = *(const f32x4*)&pf32[wave][row][c4 * 4];
            *(f32x4*)(awave + (size_t)row * SEQ + kt * 32 + c4 * 4) = v;
        }

        // --- build bf16 A-fragment of p from the f32 LDS tile ---
        v16bf pa;
        {
            const float* prow = &pf32[wave][l16][0];
            f32x4 a0 = *(const f32x4*)(prow + half * 8);
            f32x4 a1 = *(const f32x4*)(prow + half * 8 + 4);
            f32x4 a2 = *(const f32x4*)(prow + half * 8 + 16);
            f32x4 a3 = *(const f32x4*)(prow + half * 8 + 20);
#pragma unroll
            for (int i = 0; i < 4; ++i) {
                pa[i]      = (__bf16)a0[i];
                pa[4 + i]  = (__bf16)a1[i];
                pa[8 + i]  = (__bf16)a2[i];
                pa[12 + i] = (__bf16)a3[i];
            }
        }

        // --- P (16x32) @ V (32x16 B-tiles from vt): 8 WMMAs ---
#pragma unroll
        for (int dt = 0; dt < 8; ++dt) {
            v16bf vb;
            const bf16x8 lo = *(const bf16x8*)&vt[buf][dt * 16 + l16][half * 8];
            const bf16x8 hi = *(const bf16x8*)&vt[buf][dt * 16 + l16][half * 8 + 16];
            *(bf16x8*)&vb       = lo;
            *((bf16x8*)&vb + 1) = hi;
            oacc[dt] = __builtin_amdgcn_wmma_f32_16x16x32_bf16(
                false, pa, false, vb, (short)0, oacc[dt], false, false);
        }

        // --- stage next tile into the other buffer, single barrier/iter ---
        if (kt + 1 < NKT) store_tiles(buf ^ 1);

        bbase += 32 * BH;   // strength-reduced bases
        mbase += 32;

        __syncthreads();
    }

    // ---- final rowsum reduction (once, DPP16) + out = oacc / rowsum ----
    float inv[8];
#pragma unroll
    for (int j = 0; j < 8; ++j) inv[j] = 1.0f / dpp_sum16(r[j]);

#pragma unroll
    for (int dt = 0; dt < 8; ++dt) {
#pragma unroll
        for (int j = 0; j < 8; ++j) {
            const int qg = q0 + j + 8 * half;
            Out[((size_t)bh * SEQ + qg) * DIM + dt * 16 + l16] = oacc[dt][j] * inv[j];
        }
    }

    // ---- pass 2: coalesced float4 normalization of this block's attn rows ----
    if (l16 == 0) {                      // lanes 0 and 16 publish 8 rows each
#pragma unroll
        for (int j = 0; j < 8; ++j)
            invlds[wave * 16 + 8 * half + j] = inv[j];
    }
    __threadfence_block();
    __syncthreads();

    {
        const int qrow0  = qtile * 128;
        float*    block0 = Attn + ((size_t)bh * SEQ + qrow0) * SEQ;
        // 128 rows x 2048 cols = 65536 float4, 256 per thread
        for (int it = 0; it < 256; ++it) {
            const int pos = tid + it * 256;      // float4 index
            const int row = pos >> 9;            // 512 float4 per row
            const int c4  = pos & 511;
            const float iv = invlds[row];
            f32x4* ap = (f32x4*)(block0 + (size_t)row * SEQ) + c4;
            f32x4 v = *ap;
#pragma unroll
            for (int i = 0; i < 4; ++i) v[i] *= iv;
            *ap = v;
        }
    }
}

extern "C" void kernel_launch(void* const* d_in, const int* in_sizes, int n_in,
                              void* d_out, int out_size, void* d_ws, size_t ws_size,
                              hipStream_t stream) {
    const float*         Q    = (const float*)d_in[0];
    const float*         Kk   = (const float*)d_in[1];
    const float*         V    = (const float*)d_in[2];
    const float*         Bias = (const float*)d_in[3];
    const unsigned char* Mask = (const unsigned char*)d_in[4];

    float* Out  = (float*)d_out;                         // [BH, S, D]
    float* Attn = Out + (size_t)BH * SEQ * DIM;          // [BH, S, S]

    dim3 grid(BH * (SEQ / 128));   // 512 workgroups
    dim3 block(256);               // 8 wave32s; each wave owns 16 q rows
    t5_attn_kernel<<<grid, block, 0, stream>>>(Q, Kk, V, Bias, Mask, Out, Attn);
}
